// SAMV3ExemplarDetector_257698038148
// MI455X (gfx1250) — compile-verified, matrix operand order checked by reading, not run on hardware
//
#include <hip/hip_runtime.h>

// ---------------------------------------------------------------------------
// SAMV3 exemplar detector forward for MI455X (gfx1250, wave32, WMMA).
// All GEMM-shaped work runs through v_wmma_f32_16x16x32_bf16 (f32 inputs are
// converted to bf16 on the fly; accumulation stays f32).
// GEMM v3: fragment-major LDS staging -> ds_load_b128 fragment reads,
// float4 global staging -> global_load_b128, 64x64 block tile, 2 WMMA/wave,
// global_prefetch of the next K tile to hide staging latency.
// ---------------------------------------------------------------------------

constexpr int CDIM  = 256;   // model dim
constexpr int NHEAD = 8;
constexpr int HDIM  = 32;    // head dim
constexpr int ND    = 200;   // detection tokens
constexpr int NQ    = 201;   // presence + detections
constexpr int NLAY  = 6;
constexpr int MH    = 2048;  // mlp hidden
constexpr int NPE   = 128;   // C/2
constexpr int HGRID = 64, WGRID = 64;
constexpr int HW    = HGRID * WGRID;  // 4096
constexpr int BATCH = 4;
constexpr float SCALEF = 6.283185307179586f;  // 2*pi
constexpr float TEMPF  = 10000.0f;

typedef __attribute__((ext_vector_type(8)))  float          v8f;
typedef __attribute__((ext_vector_type(16))) __bf16         v16bf;
typedef __attribute__((ext_vector_type(16))) unsigned short v16u;
typedef __attribute__((ext_vector_type(8)))  unsigned short v8u;
typedef __attribute__((ext_vector_type(4)))  unsigned short v4u;

__device__ __forceinline__ unsigned short f2bf(float f) {
  unsigned int u = __builtin_bit_cast(unsigned int, f);
  unsigned int r = u + 0x7FFFu + ((u >> 16) & 1u);  // round-to-nearest-even
  return (unsigned short)(r >> 16);
}
__device__ __forceinline__ v4u pack_bf16x4(float4 f) {
  v4u r;
  r[0] = f2bf(f.x); r[1] = f2bf(f.y); r[2] = f2bf(f.z); r[3] = f2bf(f.w);
  return r;
}
__device__ __forceinline__ float sigmoidf(float x) {
  return 1.0f / (1.0f + __expf(-x));
}

// ---------------------------------------------------------------------------
// Generic batched GEMM:  Cm = alpha * A @ op(B) + bias ; optional ReLU.
//   A: M x K (lda), B: K x N (ldb) or transB -> B[n*ldb + k], C: M x N (ldc).
//   Per-z offsets: off = (z / nB2) * s?1 + (z % nB2) * s?2   (head batching).
// Block: 256 threads = 8 waves; tile 64x64; each wave owns a 16x32 strip
// (one A fragment reused across two 16x16 WMMA accumulators).
// LDS layout is fragment-major: A[m][k], Bt[n][k], rows padded to 80 B so a
// lane's 16 bf16 fragment elements are two aligned 16 B chunks (ds_load_b128).
// ---------------------------------------------------------------------------
#define BM 64
#define BN 64
#define BKK 32
#define LDSP 40  // padded row pitch in ushorts (80 B, keeps 16 B alignment)

__global__ void __launch_bounds__(256) gemm_bf16_wmma(
    const float* __restrict__ A, const float* __restrict__ B,
    const float* __restrict__ bias, float* __restrict__ Cm,
    int M, int N, int K, int lda, int ldb, int ldc,
    int nB2, long sA1, long sA2, long sB1, long sB2, long sC1, long sC2,
    int transB, int relu, float alpha)
{
  __shared__ alignas(16) unsigned short shA[BM][LDSP];
  __shared__ alignas(16) unsigned short shB[BN][LDSP];  // B transposed: [n][k]

  const int tid  = threadIdx.x;
  const int lane = tid & 31;
  const int wave = tid >> 5;       // 8 waves
  const int wm   = wave >> 1;      // 0..3  (M sub-tile)
  const int wn   = wave & 1;       // 0..1  (32-col strip)
  const int l16  = lane & 15;
  const int half = lane >> 4;
  const int row0 = blockIdx.y * BM;
  const int col0 = blockIdx.x * BN;
  const int z    = blockIdx.z;

  A  += (z / nB2) * sA1 + (z % nB2) * sA2;
  B  += (z / nB2) * sB1 + (z % nB2) * sB2;
  Cm += (z / nB2) * sC1 + (z % nB2) * sC2;

  const bool fullM = (row0 + BM) <= M;
  const bool fullN = (col0 + BN) <= N;

  v8f acc0 = {};
  v8f acc1 = {};

  for (int k0 = 0; k0 < K; k0 += BKK) {
    const bool fullK = (k0 + BKK) <= K;
    const bool nextK = (k0 + 2 * BKK) <= K;

    // ---- stage A tile (64x32): each thread packs 2x 4 consecutive k -------
    if (fullM && fullK && ((lda & 3) == 0)) {
#pragma unroll
      for (int i = 0; i < 2; ++i) {
        int r  = (tid >> 3) + i * 32;
        int cb = (tid & 7) * 4;
        const float* ap = A + (long)(row0 + r) * lda + (k0 + cb);
        float4 f = *(const float4*)ap;
        *(v4u*)&shA[r][cb] = pack_bf16x4(f);
        if (nextK) __builtin_prefetch(ap + BKK, 0, 3);  // global_prefetch next tile
      }
    } else if (fullM && fullK) {
#pragma unroll
      for (int i = 0; i < 2; ++i) {
        int r  = (tid >> 3) + i * 32;
        int cb = (tid & 7) * 4;
        const float* ap = A + (long)(row0 + r) * lda + (k0 + cb);
        v4u p;
#pragma unroll
        for (int j = 0; j < 4; ++j) p[j] = f2bf(ap[j]);
        *(v4u*)&shA[r][cb] = p;
      }
    } else {
#pragma unroll
      for (int i = 0; i < 2; ++i) {
        int r  = (tid >> 3) + i * 32;
        int cb = (tid & 7) * 4;
        bool rok = (row0 + r) < M;
        v4u p;
#pragma unroll
        for (int j = 0; j < 4; ++j) {
          float v = 0.0f;
          if (rok && (k0 + cb + j) < K) v = A[(long)(row0 + r) * lda + (k0 + cb + j)];
          p[j] = f2bf(v);
        }
        *(v4u*)&shA[r][cb] = p;
      }
    }

    // ---- stage B tile into Bt[n][k] (64 n x 32 k) --------------------------
    if (transB) {
      // global B is [n][k]: contiguous in k -> float4 loads
      if (fullN && fullK && ((ldb & 3) == 0)) {
#pragma unroll
        for (int i = 0; i < 2; ++i) {
          int n  = (tid >> 3) + i * 32;
          int kb = (tid & 7) * 4;
          const float* bp = B + (long)(col0 + n) * ldb + (k0 + kb);
          float4 f = *(const float4*)bp;
          *(v4u*)&shB[n][kb] = pack_bf16x4(f);
          if (nextK) __builtin_prefetch(bp + BKK, 0, 3);
        }
      } else {
#pragma unroll
        for (int i = 0; i < 2; ++i) {
          int n  = (tid >> 3) + i * 32;
          int kb = (tid & 7) * 4;
          bool nok = (col0 + n) < N;
          v4u p;
#pragma unroll
          for (int j = 0; j < 4; ++j) {
            float v = 0.0f;
            if (nok && (k0 + kb + j) < K) v = B[(long)(col0 + n) * ldb + (k0 + kb + j)];
            p[j] = f2bf(v);
          }
          *(v4u*)&shB[n][kb] = p;
        }
      }
    } else {
      // global B is [k][n]: lanes sweep n (coalesced), thread owns 4 k's
      if (fullN && fullK) {
#pragma unroll
        for (int i = 0; i < 2; ++i) {
          int n  = tid & 63;
          int kb = (tid >> 6) * 4 + i * 16;
          v4u p;
#pragma unroll
          for (int j = 0; j < 4; ++j)
            p[j] = f2bf(B[(long)(k0 + kb + j) * ldb + (col0 + n)]);
          *(v4u*)&shB[n][kb] = p;
          if (nextK) __builtin_prefetch(B + (long)(k0 + BKK + kb) * ldb + (col0 + n), 0, 3);
        }
      } else {
#pragma unroll
        for (int i = 0; i < 2; ++i) {
          int n  = tid & 63;
          int kb = (tid >> 6) * 4 + i * 16;
          bool nok = (col0 + n) < N;
          v4u p;
#pragma unroll
          for (int j = 0; j < 4; ++j) {
            float v = 0.0f;
            if (nok && (k0 + kb + j) < K) v = B[(long)(k0 + kb + j) * ldb + (col0 + n)];
            p[j] = f2bf(v);
          }
          *(v4u*)&shB[n][kb] = p;
        }
      }
    }
    __syncthreads();

    // ---- fragment reads: 2x ds_load_b128 each ------------------------------
    // CDNA5 16-bit A layout: lane(0-15)->M, VGPR v holds K pairs; lane half
    // selects K blocks {half*8..+7, 16+half*8..+7} -> contiguous in our LDS.
    const int arow = wm * 16 + l16;
    v8u alo = *(const v8u*)&shA[arow][half * 8];
    v8u ahi = *(const v8u*)&shA[arow][16 + half * 8];
    v16u au = __builtin_shufflevector(alo, ahi, 0, 1, 2, 3, 4, 5, 6, 7,
                                      8, 9, 10, 11, 12, 13, 14, 15);
    const int brow0 = wn * 32 + l16;
    v8u blo0 = *(const v8u*)&shB[brow0][half * 8];
    v8u bhi0 = *(const v8u*)&shB[brow0][16 + half * 8];
    v16u bu0 = __builtin_shufflevector(blo0, bhi0, 0, 1, 2, 3, 4, 5, 6, 7,
                                       8, 9, 10, 11, 12, 13, 14, 15);
    const int brow1 = brow0 + 16;
    v8u blo1 = *(const v8u*)&shB[brow1][half * 8];
    v8u bhi1 = *(const v8u*)&shB[brow1][16 + half * 8];
    v16u bu1 = __builtin_shufflevector(blo1, bhi1, 0, 1, 2, 3, 4, 5, 6, 7,
                                       8, 9, 10, 11, 12, 13, 14, 15);

    acc0 = __builtin_amdgcn_wmma_f32_16x16x32_bf16(
        false, __builtin_bit_cast(v16bf, au),
        false, __builtin_bit_cast(v16bf, bu0),
        (short)0, acc0, false, false);
    acc1 = __builtin_amdgcn_wmma_f32_16x16x32_bf16(
        false, __builtin_bit_cast(v16bf, au),
        false, __builtin_bit_cast(v16bf, bu1),
        (short)0, acc1, false, false);
    __syncthreads();
  }

  // ---- writeback: C/D f32 layout lane(0-15)->N, VGPR v->M (+8 upper half) --
#pragma unroll
  for (int vv = 0; vv < 8; ++vv) {
    int r  = row0 + wm * 16 + half * 8 + vv;
    int c0 = col0 + wn * 32 + l16;
    int c1 = c0 + 16;
    if (r < M && c0 < N) {
      float o = acc0[vv] * alpha + (bias ? bias[c0] : 0.0f);
      if (relu) o = fmaxf(o, 0.0f);
      Cm[(long)r * ldc + c0] = o;
    }
    if (r < M && c1 < N) {
      float o = acc1[vv] * alpha + (bias ? bias[c1] : 0.0f);
      if (relu) o = fmaxf(o, 0.0f);
      Cm[(long)r * ldc + c1] = o;
    }
  }
}

// ---------------------------------------------------------------------------
// Elementwise / reduction kernels
// ---------------------------------------------------------------------------
__global__ void transpose_img_kernel(const float* __restrict__ in, float* __restrict__ out) {
  long i = (long)blockIdx.x * 256 + threadIdx.x;  // over B*HW*CDIM
  if (i >= (long)BATCH * HW * CDIM) return;
  int c = (int)(i % CDIM);
  long t = i / CDIM;
  int p = (int)(t % HW);
  int b = (int)(t / HW);
  out[i] = in[((long)b * CDIM + c) * HW + p];
}

__global__ void imgpos_kernel(float* __restrict__ pos) {
  int i = blockIdx.x * 256 + threadIdx.x;  // HW*CDIM
  if (i >= HW * CDIM) return;
  int c = i % CDIM, p = i / CDIM;
  int y = p >> 6, x = p & 63;
  float v; int ii;
  if (c < NPE) { v = (y + 0.5f) / (float)HGRID; ii = c; }
  else         { v = (x + 0.5f) / (float)WGRID; ii = c - NPE; }
  float dim_t = powf(TEMPF, 2.0f * (float)(ii >> 1) / (float)NPE);
  float a = v * SCALEF / dim_t;
  pos[i] = (ii & 1) ? cosf(a) : sinf(a);
}

__global__ void addpos_kernel(const float* __restrict__ tok, const float* __restrict__ pos,
                              float* __restrict__ out) {
  long i = (long)blockIdx.x * 256 + threadIdx.x;
  if (i >= (long)BATCH * HW * CDIM) return;
  out[i] = tok[i] + pos[i % ((long)HW * CDIM)];
}

__global__ void add_kernel(const float* __restrict__ a, const float* __restrict__ b,
                           float* __restrict__ o, long n) {
  long i = (long)blockIdx.x * 256 + threadIdx.x;
  if (i < n) o[i] = a[i] + b[i];
}

__global__ void init_q_kernel(const float* __restrict__ det_tok, const float* __restrict__ pres_tok,
                              float* __restrict__ q) {
  int i = blockIdx.x * 256 + threadIdx.x;  // B*NQ*CDIM
  if (i >= BATCH * NQ * CDIM) return;
  int c = i % CDIM;
  int t = (i / CDIM) % NQ;
  q[i] = (t == 0) ? pres_tok[c] : det_tok[(t - 1) * CDIM + c];
}

__global__ void build_qpos_kernel(const float* __restrict__ det_pos, float* __restrict__ qpos) {
  int i = blockIdx.x * 256 + threadIdx.x;  // B*NQ*CDIM
  if (i >= BATCH * NQ * CDIM) return;
  int c = i % CDIM;
  int t = (i / CDIM) % NQ;
  int b = i / (NQ * CDIM);
  qpos[i] = (t == 0) ? 0.0f : det_pos[((long)(b * ND) + (t - 1)) * CDIM + c];
}

__global__ void anchor_init_kernel(const float* __restrict__ a0, float* __restrict__ anc) {
  int i = blockIdx.x * 256 + threadIdx.x;
  if (i >= BATCH * ND * 4) return;
  anc[i] = a0[i % (ND * 4)];
}

__global__ void box_kernel(const float* __restrict__ anc, float* __restrict__ box_c,
                           float* __restrict__ box_xyxy) {
  int i = blockIdx.x * 256 + threadIdx.x;  // B*ND
  if (i >= BATCH * ND) return;
  const float* a = anc + (long)i * 4;
  float cx = sigmoidf(a[0]), cy = sigmoidf(a[1]);
  float w  = sigmoidf(a[2]), h  = sigmoidf(a[3]);
  box_c[i * 4 + 0] = cx; box_c[i * 4 + 1] = cy;
  box_c[i * 4 + 2] = w;  box_c[i * 4 + 3] = h;
  box_xyxy[i * 4 + 0] = cx - 0.5f * w;
  box_xyxy[i * 4 + 1] = cy - 0.5f * h;
  box_xyxy[i * 4 + 2] = cx + 0.5f * w;
  box_xyxy[i * 4 + 3] = cy + 0.5f * h;
}

__global__ void posembed_kernel(const float* __restrict__ box_c, float* __restrict__ out) {
  int i = blockIdx.x * 256 + threadIdx.x;  // B*ND*2C
  if (i >= BATCH * ND * 2 * CDIM) return;
  int c = i % (2 * CDIM);
  int bn = i / (2 * CDIM);
  int g = c / NPE;                        // group order y,x,w,h
  int ii = c % NPE;
  int sel = (g == 0) ? 1 : (g == 1) ? 0 : g;
  float v = box_c[(long)bn * 4 + sel];
  float dim_t = powf(TEMPF, 2.0f * (float)(ii >> 1) / (float)NPE);
  float a = v * SCALEF / dim_t;
  out[i] = (ii & 1) ? cosf(a) : sinf(a);
}

__global__ void relin_kernel(const float* __restrict__ box_xyxy,
                             float* __restrict__ dxin, float* __restrict__ dyin) {
  int i = blockIdx.x * 256 + threadIdx.x;  // B*ND*64*2
  if (i >= BATCH * ND * 64 * 2) return;
  int j = i & 1;
  int t = i >> 1;
  int x = t & 63;
  int bn = t >> 6;
  float tok = (x + 0.5f) / 64.0f;
  dxin[i] = tok - box_xyxy[(long)bn * 4 + (j ? 2 : 0)];
  dyin[i] = tok - box_xyxy[(long)bn * 4 + (j ? 3 : 1)];
}

// softmax over row of length Nk; optional rel-pos bias (image attn) and
// exemplar key-mask. grid = (NQ, NHEAD, BATCH), block = 256.
__global__ void __launch_bounds__(256) softmax_kernel(
    float* __restrict__ scores, int Nk,
    const float* __restrict__ bx, const float* __restrict__ by,
    const unsigned char* __restrict__ mask)
{
  int qi = blockIdx.x, h = blockIdx.y, b = blockIdx.z;
  float* row = scores + ((long)(b * NHEAD + h) * NQ + qi) * Nk;
  __shared__ float sh[256];
  int tid = threadIdx.x;

  float mx = -1e30f;
  for (int k = tid; k < Nk; k += 256) {
    float v = row[k];
    if (bx != nullptr && qi > 0) {
      int n = qi - 1, y = k >> 6, x = k & 63;
      v += by[(((long)b * ND + n) * 64 + y) * NHEAD + h]
         + bx[(((long)b * ND + n) * 64 + x) * NHEAD + h];
    }
    if (mask != nullptr && mask[b * 8 + k]) v = -1e9f;
    row[k] = v;
    mx = fmaxf(mx, v);
  }
  sh[tid] = mx; __syncthreads();
  for (int s = 128; s > 0; s >>= 1) { if (tid < s) sh[tid] = fmaxf(sh[tid], sh[tid + s]); __syncthreads(); }
  mx = sh[0]; __syncthreads();

  float sum = 0.0f;
  for (int k = tid; k < Nk; k += 256) { float e = __expf(row[k] - mx); row[k] = e; sum += e; }
  sh[tid] = sum; __syncthreads();
  for (int s = 128; s > 0; s >>= 1) { if (tid < s) sh[tid] += sh[tid + s]; __syncthreads(); }
  float inv = 1.0f / sh[0];
  for (int k = tid; k < Nk; k += 256) row[k] *= inv;
}

// LayerNorm over CDIM with optional residual; grid=(rows_per_batch, BATCH).
__global__ void __launch_bounds__(256) ln_kernel(
    const float* __restrict__ x, const float* __restrict__ res, float* __restrict__ out,
    const float* __restrict__ w, const float* __restrict__ bb,
    long xbstride, long obstride, int rpb)
{
  int r = blockIdx.x, b = blockIdx.y, tid = threadIdx.x;
  float v = x[(long)b * xbstride + (long)r * CDIM + tid];
  if (res != nullptr) v += res[((long)b * rpb + r) * CDIM + tid];
  __shared__ float sh[256];
  sh[tid] = v; __syncthreads();
  for (int s = 128; s > 0; s >>= 1) { if (tid < s) sh[tid] += sh[tid + s]; __syncthreads(); }
  float mean = sh[0] * (1.0f / CDIM); __syncthreads();
  float d = v - mean;
  sh[tid] = d * d; __syncthreads();
  for (int s = 128; s > 0; s >>= 1) { if (tid < s) sh[tid] += sh[tid + s]; __syncthreads(); }
  float var = sh[0] * (1.0f / CDIM);
  out[(long)b * obstride + (long)r * CDIM + tid] = d * rsqrtf(var + 1e-5f) * w[tid] + bb[tid];
}

__global__ void pool_kernel(const float* __restrict__ ex, const unsigned char* __restrict__ mask,
                            float* __restrict__ pooled) {
  int b = blockIdx.x, c = threadIdx.x;
  float s = 0.0f, cnt = 0.0f;
  for (int n = 0; n < 8; ++n) {
    float valid = mask[b * 8 + n] ? 0.0f : 1.0f;
    s += ex[((long)b * 8 + n) * CDIM + c] * valid;
    cnt += valid;
  }
  pooled[b * CDIM + c] = s / fmaxf(cnt, 1.0f);
}

__global__ void pres_kernel(const float* __restrict__ p1, float* __restrict__ out_pres) {
  int b = threadIdx.x;
  if (b < BATCH) out_pres[b] = sigmoidf(p1[b]);
}

__global__ void __launch_bounds__(256) detscore_kernel(
    const float* __restrict__ det_out, const float* __restrict__ z,
    const float* __restrict__ pres, float* __restrict__ out) {
  int n = blockIdx.x, b = blockIdx.y, tid = threadIdx.x;
  __shared__ float sh[256];
  sh[tid] = det_out[((long)b * ND + n) * CDIM + tid] * z[b * CDIM + tid];
  __syncthreads();
  for (int s = 128; s > 0; s >>= 1) { if (tid < s) sh[tid] += sh[tid + s]; __syncthreads(); }
  if (tid == 0) out[b * ND + n] = pres[b] * sigmoidf(sh[0] * (1.0f / 16.0f));
}

// ---------------------------------------------------------------------------
// Host orchestration
// ---------------------------------------------------------------------------
static inline void gemmB(hipStream_t s,
    const float* A, const float* B, const float* bias, float* Cm,
    int M, int N, int K, int lda, int ldb, int ldc,
    int batch, int nB2,
    long sA1, long sA2, long sB1, long sB2, long sC1, long sC2,
    int transB, int relu, float alpha) {
  dim3 g((N + BN - 1) / BN, (M + BM - 1) / BM, batch);
  gemm_bf16_wmma<<<g, dim3(256), 0, s>>>(A, B, bias, Cm, M, N, K, lda, ldb, ldc,
                                         nB2, sA1, sA2, sB1, sB2, sC1, sC2,
                                         transB, relu, alpha);
}
static inline void gemmP(hipStream_t s, const float* A, const float* B, const float* bias,
                         float* Cm, int M, int N, int K, int relu, float alpha = 1.0f) {
  gemmB(s, A, B, bias, Cm, M, N, K, K, N, N, 1, 1, 0, 0, 0, 0, 0, 0, 0, relu, alpha);
}

static inline float* wsf(void* base, size_t& off, size_t nfloats) {
  float* p = (float*)((char*)base + off);
  off = (off + nfloats * sizeof(float) + 255) & ~(size_t)255;
  return p;
}

extern "C" void kernel_launch(void* const* d_in, const int* in_sizes, int n_in,
                              void* d_out, int out_size, void* d_ws, size_t ws_size,
                              hipStream_t stream) {
  (void)in_sizes; (void)n_in; (void)out_size; (void)ws_size;

  const float* img = (const float*)d_in[0];                 // (B,C,64,64)
  const float* exm = (const float*)d_in[1];                 // (B,8,C)
  const unsigned char* mask = (const unsigned char*)d_in[2];// (B,8) bool
  // params pytree leaves (jax tree_flatten: dict keys alphabetical)
  const float* const* P = (const float* const*)(d_in + 3);

  // leaf map: 0 anchor_boxes, 1 detection_tokens,
  // 2..217 layers (36/layer: ex[10], im[10], mlp[6], sa[10];
  //   attn order: bk,bo,bq,bv,ln_b,ln_w,wk,wo,wq,wv ; mlp: b1,b2,ln_b,ln_w,w1,w2)
  auto AT = [&](int layer, int base, int k) { return P[2 + layer * 36 + base + k]; };
  const float* anchor0  = P[0];
  const float* det_tok  = P[1];
  const float *d2b_b0 = P[218], *d2b_w0 = P[219], *d2b_b1 = P[220], *d2b_w1 = P[221],
              *d2b_b2 = P[222], *d2b_w2 = P[223];
  const float *pe_b0 = P[224], *pe_w0 = P[225], *pe_b1 = P[226], *pe_w1 = P[227];
  const float *dx_b0 = P[228], *dx_w0 = P[229], *dx_b1 = P[230], *dx_w1 = P[231];
  const float *dy_b0 = P[232], *dy_w0 = P[233], *dy_b1 = P[234], *dy_w1 = P[235];
  const float *pr_b0 = P[236], *pr_w0 = P[237], *pr_b1 = P[238], *pr_w1 = P[239];
  const float *oln_b = P[240], *oln_w = P[241];
  const float* pres_tok = P[242];
  const float *sc_b0 = P[243], *sc_w0 = P[244], *sc_b1 = P[245], *sc_w1 = P[246];

  // workspace
  size_t off = 0;
  float* img_tok = wsf(d_ws, off, (size_t)BATCH * HW * CDIM);
  float* img_pos = wsf(d_ws, off, (size_t)HW * CDIM);
  float* img_kin = wsf(d_ws, off, (size_t)BATCH * HW * CDIM);
  float* imgK    = wsf(d_ws, off, (size_t)BATCH * HW * CDIM);
  float* imgV    = wsf(d_ws, off, (size_t)BATCH * HW * CDIM);
  float* q       = wsf(d_ws, off, (size_t)BATCH * NQ * CDIM);
  float* qpos    = wsf(d_ws, off, (size_t)BATCH * NQ * CDIM);
  float* qq      = wsf(d_ws, off, (size_t)BATCH * NQ * CDIM);
  float* Qp      = wsf(d_ws, off, (size_t)BATCH * NQ * CDIM);
  float* KpS     = wsf(d_ws, off, (size_t)BATCH * NQ * CDIM);
  float* VpS     = wsf(d_ws, off, (size_t)BATCH * NQ * CDIM);
  float* KpE     = wsf(d_ws, off, (size_t)BATCH * 8 * CDIM);
  float* VpE     = wsf(d_ws, off, (size_t)BATCH * 8 * CDIM);
  float* det_pos = wsf(d_ws, off, (size_t)BATCH * ND * CDIM);
  float* attn_o  = wsf(d_ws, off, (size_t)BATCH * NQ * CDIM);
  float* mha     = wsf(d_ws, off, (size_t)BATCH * NQ * CDIM);
  float* h1      = wsf(d_ws, off, (size_t)BATCH * NQ * MH);
  float* scores  = wsf(d_ws, off, (size_t)BATCH * NHEAD * NQ * HW);
  float* anc     = wsf(d_ws, off, (size_t)BATCH * ND * 4);
  float* box_c   = wsf(d_ws, off, (size_t)BATCH * ND * 4);
  float* box_xy  = wsf(d_ws, off, (size_t)BATCH * ND * 4);
  float* dxin    = wsf(d_ws, off, (size_t)BATCH * ND * 64 * 2);
  float* dyin    = wsf(d_ws, off, (size_t)BATCH * ND * 64 * 2);
  float* relh    = wsf(d_ws, off, (size_t)BATCH * ND * 64 * CDIM);
  float* bx      = wsf(d_ws, off, (size_t)BATCH * ND * 64 * NHEAD);
  float* by      = wsf(d_ws, off, (size_t)BATCH * ND * 64 * NHEAD);
  float* posy    = wsf(d_ws, off, (size_t)BATCH * ND * 2 * CDIM);
  float* delta   = wsf(d_ws, off, (size_t)BATCH * ND * 4);
  float* dethid  = wsf(d_ws, off, (size_t)BATCH * ND * CDIM);
  float* dethid2 = wsf(d_ws, off, (size_t)BATCH * ND * CDIM);
  float* pooled  = wsf(d_ws, off, (size_t)BATCH * CDIM);
  float* sc_h    = wsf(d_ws, off, (size_t)BATCH * MH);
  float* zvec    = wsf(d_ws, off, (size_t)BATCH * CDIM);
  float* ph      = wsf(d_ws, off, (size_t)BATCH * CDIM);
  float* p1      = wsf(d_ws, off, (size_t)BATCH);

  float* out_det  = (float*)d_out;                 // B*ND*C
  float* out_box  = out_det + (size_t)BATCH * ND * CDIM;
  float* out_sc   = out_box + (size_t)BATCH * ND * 4;
  float* out_pres = out_sc + (size_t)BATCH * ND;

  const long nQC  = (long)BATCH * NQ * CDIM;
  const long nImg = (long)BATCH * HW * CDIM;
  const float iscl = 0.17677669529663689f;  // 1/sqrt(32)

  // ---- prologue -----------------------------------------------------------
  transpose_img_kernel<<<(int)((nImg + 255) / 256), 256, 0, stream>>>(img, img_tok);
  imgpos_kernel<<<(HW * CDIM + 255) / 256, 256, 0, stream>>>(img_pos);
  addpos_kernel<<<(int)((nImg + 255) / 256), 256, 0, stream>>>(img_tok, img_pos, img_kin);
  init_q_kernel<<<(BATCH * NQ * CDIM + 255) / 256, 256, 0, stream>>>(det_tok, pres_tok, q);
  anchor_init_kernel<<<(BATCH * ND * 4 + 255) / 256, 256, 0, stream>>>(anchor0, anc);

  for (int l = 0; l < NLAY; ++l) {
    const int EX = 0, IM = 10, MLPB = 20, SA = 26;
    // attn leaf order: 0 bk,1 bo,2 bq,3 bv,4 ln_b,5 ln_w,6 wk,7 wo,8 wq,9 wv
    // ---- box geometry + positional encodings ------------------------------
    box_kernel<<<(BATCH * ND + 255) / 256, 256, 0, stream>>>(anc, box_c, box_xy);
    posembed_kernel<<<(BATCH * ND * 2 * CDIM + 255) / 256, 256, 0, stream>>>(box_c, posy);
    gemmP(stream, posy, pe_w0, pe_b0, dethid, BATCH * ND, CDIM, 2 * CDIM, 1);
    gemmP(stream, dethid, pe_w1, pe_b1, det_pos, BATCH * ND, CDIM, CDIM, 0);
    build_qpos_kernel<<<(BATCH * NQ * CDIM + 255) / 256, 256, 0, stream>>>(det_pos, qpos);
    // rel-pos bias MLPs
    relin_kernel<<<(BATCH * ND * 128 + 255) / 256, 256, 0, stream>>>(box_xy, dxin, dyin);
    gemmP(stream, dxin, dx_w0, dx_b0, relh, BATCH * ND * 64, CDIM, 2, 1);
    gemmP(stream, relh, dx_w1, dx_b1, bx, BATCH * ND * 64, NHEAD, CDIM, 0);
    gemmP(stream, dyin, dy_w0, dy_b0, relh, BATCH * ND * 64, CDIM, 2, 1);
    gemmP(stream, relh, dy_w1, dy_b1, by, BATCH * ND * 64, NHEAD, CDIM, 0);

    // ---- self-attention ---------------------------------------------------
    add_kernel<<<(int)((nQC + 255) / 256), 256, 0, stream>>>(q, qpos, qq, nQC);
    gemmP(stream, qq, AT(l, SA, 8), AT(l, SA, 2), Qp, BATCH * NQ, CDIM, CDIM, 0);
    gemmP(stream, qq, AT(l, SA, 6), AT(l, SA, 0), KpS, BATCH * NQ, CDIM, CDIM, 0);
    gemmP(stream, q,  AT(l, SA, 9), AT(l, SA, 3), VpS, BATCH * NQ, CDIM, CDIM, 0);
    gemmB(stream, Qp, KpS, nullptr, scores, NQ, NQ, HDIM, CDIM, CDIM, NQ,
          BATCH * NHEAD, NHEAD, (long)NQ * CDIM, HDIM, (long)NQ * CDIM, HDIM,
          (long)NHEAD * NQ * NQ, (long)NQ * NQ, 1, 0, iscl);
    softmax_kernel<<<dim3(NQ, NHEAD, BATCH), 256, 0, stream>>>(scores, NQ, nullptr, nullptr, nullptr);
    gemmB(stream, scores, VpS, nullptr, attn_o, NQ, HDIM, NQ, NQ, CDIM, CDIM,
          BATCH * NHEAD, NHEAD, (long)NHEAD * NQ * NQ, (long)NQ * NQ,
          (long)NQ * CDIM, HDIM, (long)NQ * CDIM, HDIM, 0, 0, 1.0f);
    gemmP(stream, attn_o, AT(l, SA, 7), AT(l, SA, 1), mha, BATCH * NQ, CDIM, CDIM, 0);
    ln_kernel<<<dim3(NQ, BATCH), 256, 0, stream>>>(q, mha, q, AT(l, SA, 5), AT(l, SA, 4),
                                                   (long)NQ * CDIM, (long)NQ * CDIM, NQ);

    // ---- exemplar cross-attention ----------------------------------------
    add_kernel<<<(int)((nQC + 255) / 256), 256, 0, stream>>>(q, qpos, qq, nQC);
    gemmP(stream, qq,  AT(l, EX, 8), AT(l, EX, 2), Qp, BATCH * NQ, CDIM, CDIM, 0);
    gemmP(stream, exm, AT(l, EX, 6), AT(l, EX, 0), KpE, BATCH * 8, CDIM, CDIM, 0);
    gemmP(stream, exm, AT(l, EX, 9), AT(l, EX, 3), VpE, BATCH * 8, CDIM, CDIM, 0);
    gemmB(stream, Qp, KpE, nullptr, scores, NQ, 8, HDIM, CDIM, CDIM, 8,
          BATCH * NHEAD, NHEAD, (long)NQ * CDIM, HDIM, (long)8 * CDIM, HDIM,
          (long)NHEAD * NQ * 8, (long)NQ * 8, 1, 0, iscl);
    softmax_kernel<<<dim3(NQ, NHEAD, BATCH), 256, 0, stream>>>(scores, 8, nullptr, nullptr, mask);
    gemmB(stream, scores, VpE, nullptr, attn_o, NQ, HDIM, 8, 8, CDIM, CDIM,
          BATCH * NHEAD, NHEAD, (long)NHEAD * NQ * 8, (long)NQ * 8,
          (long)8 * CDIM, HDIM, (long)NQ * CDIM, HDIM, 0, 0, 1.0f);
    gemmP(stream, attn_o, AT(l, EX, 7), AT(l, EX, 1), mha, BATCH * NQ, CDIM, CDIM, 0);
    ln_kernel<<<dim3(NQ, BATCH), 256, 0, stream>>>(q, mha, q, AT(l, EX, 5), AT(l, EX, 4),
                                                   (long)NQ * CDIM, (long)NQ * CDIM, NQ);

    // ---- image cross-attention (with rel-pos bias) ------------------------
    add_kernel<<<(int)((nQC + 255) / 256), 256, 0, stream>>>(q, qpos, qq, nQC);
    gemmP(stream, qq, AT(l, IM, 8), AT(l, IM, 2), Qp, BATCH * NQ, CDIM, CDIM, 0);
    gemmP(stream, img_kin, AT(l, IM, 6), AT(l, IM, 0), imgK, BATCH * HW, CDIM, CDIM, 0);
    gemmP(stream, img_tok, AT(l, IM, 9), AT(l, IM, 3), imgV, BATCH * HW, CDIM, CDIM, 0);
    gemmB(stream, Qp, imgK, nullptr, scores, NQ, HW, HDIM, CDIM, CDIM, HW,
          BATCH * NHEAD, NHEAD, (long)NQ * CDIM, HDIM, (long)HW * CDIM, HDIM,
          (long)NHEAD * NQ * HW, (long)NQ * HW, 1, 0, iscl);
    softmax_kernel<<<dim3(NQ, NHEAD, BATCH), 256, 0, stream>>>(scores, HW, bx, by, nullptr);
    gemmB(stream, scores, imgV, nullptr, attn_o, NQ, HDIM, HW, HW, CDIM, CDIM,
          BATCH * NHEAD, NHEAD, (long)NHEAD * NQ * HW, (long)NQ * HW,
          (long)HW * CDIM, HDIM, (long)NQ * CDIM, HDIM, 0, 0, 1.0f);
    gemmP(stream, attn_o, AT(l, IM, 7), AT(l, IM, 1), mha, BATCH * NQ, CDIM, CDIM, 0);
    ln_kernel<<<dim3(NQ, BATCH), 256, 0, stream>>>(q, mha, q, AT(l, IM, 5), AT(l, IM, 4),
                                                   (long)NQ * CDIM, (long)NQ * CDIM, NQ);

    // ---- feed-forward -----------------------------------------------------
    gemmP(stream, q, AT(l, MLPB, 4), AT(l, MLPB, 0), h1, BATCH * NQ, MH, CDIM, 1);
    gemmP(stream, h1, AT(l, MLPB, 5), AT(l, MLPB, 1), mha, BATCH * NQ, CDIM, MH, 0);
    ln_kernel<<<dim3(NQ, BATCH), 256, 0, stream>>>(q, mha, q, AT(l, MLPB, 3), AT(l, MLPB, 2),
                                                   (long)NQ * CDIM, (long)NQ * CDIM, NQ);

    // ---- anchor refinement (det rows are q[:,1:], strided per batch) ------
    gemmB(stream, q + CDIM, d2b_w0, d2b_b0, dethid, ND, CDIM, CDIM, CDIM, CDIM, CDIM,
          BATCH, 1, (long)NQ * CDIM, 0, 0, 0, (long)ND * CDIM, 0, 0, 1, 1.0f);
    gemmP(stream, dethid, d2b_w1, d2b_b1, dethid2, BATCH * ND, CDIM, CDIM, 1);
    gemmP(stream, dethid2, d2b_w2, d2b_b2, delta, BATCH * ND, 4, CDIM, 0);
    add_kernel<<<(BATCH * ND * 4 + 255) / 256, 256, 0, stream>>>(anc, delta, anc,
                                                                 (long)BATCH * ND * 4);
  }

  // ---- heads --------------------------------------------------------------
  box_kernel<<<(BATCH * ND + 255) / 256, 256, 0, stream>>>(anc, box_c, out_box);
  ln_kernel<<<dim3(ND, BATCH), 256, 0, stream>>>(q + CDIM, nullptr, out_det, oln_w, oln_b,
                                                 (long)NQ * CDIM, (long)ND * CDIM, ND);
  // presence head on q[:,0]
  gemmB(stream, q, pr_w0, pr_b0, ph, 1, CDIM, CDIM, CDIM, CDIM, CDIM,
        BATCH, 1, (long)NQ * CDIM, 0, 0, 0, (long)CDIM, 0, 0, 1, 1.0f);
  gemmP(stream, ph, pr_w1, pr_b1, p1, BATCH, 1, CDIM, 0);
  pres_kernel<<<1, 32, 0, stream>>>(p1, out_pres);
  // exemplar pooling + scoring
  pool_kernel<<<BATCH, 256, 0, stream>>>(exm, mask, pooled);
  gemmP(stream, pooled, sc_w0, sc_b0, sc_h, BATCH, MH, CDIM, 1);
  gemmP(stream, sc_h, sc_w1, sc_b1, zvec, BATCH, CDIM, MH, 0);
  detscore_kernel<<<dim3(ND, BATCH), 256, 0, stream>>>(out_det, zvec, out_pres, out_sc);
}
